// GNNLayerKAFP_76871324663923
// MI455X (gfx1250) — matrix operand order repaired and checked by reading, not compiled, not run on hardware
//
#include <hip/hip_runtime.h>
#include <hip/hip_bf16.h>

typedef __attribute__((ext_vector_type(16))) _Float16 v16h;
typedef __attribute__((ext_vector_type(8)))  float    v8f;

#define V_N 100000
#define E_N 400000
#define D_N 128
#define KP_N 20
#define KRON_K 400
#define KRON_KP 416
#define KRON_KS (KRON_KP / 32)   // 13 k-steps
#define EPS 1e-5f

// ---------------- utility kernels ----------------

__global__ void zero_kernel(float* __restrict__ p, long n) {
    long i = (long)blockIdx.x * blockDim.x + threadIdx.x;
    if (i < n) p[i] = 0.f;
}

// dst[n*Kpad+k] = (k<K) ? (transpose ? src[k*N+n] : src[n*K+k]) : 0   (f32 -> f16)
__global__ void cvt_wt_kernel(_Float16* __restrict__ dst, const float* __restrict__ src,
                              int K, int N, int Kpad, int transpose) {
    long i = (long)blockIdx.x * blockDim.x + threadIdx.x;
    if (i >= (long)N * Kpad) return;
    int n = (int)(i / Kpad);
    int k = (int)(i % Kpad);
    float v = 0.f;
    if (k < K) v = transpose ? src[(long)k * N + n] : src[(long)n * K + k];
    dst[i] = (_Float16)v;
}

__global__ void relu_kernel(float* __restrict__ p, long n) {
    long i = (long)blockIdx.x * blockDim.x + threadIdx.x;
    if (i < n) p[i] = fmaxf(p[i], 0.f);
}

// ---------------- node projection to KP=20 + LN + relu ----------------

__global__ void npj_kernel(const float* __restrict__ nf, const float* __restrict__ Wk1,
                           const float* __restrict__ bk1, const float* __restrict__ g1,
                           const float* __restrict__ b1, float* __restrict__ npj) {
    int n = blockIdx.x * blockDim.x + threadIdx.x;
    if (n >= V_N) return;
    float acc[KP_N];
#pragma unroll
    for (int j = 0; j < KP_N; ++j) acc[j] = bk1[j];
    for (int i = 0; i < D_N; ++i) {
        float x = nf[(long)n * D_N + i];
#pragma unroll
        for (int j = 0; j < KP_N; ++j) acc[j] += x * Wk1[i * KP_N + j];
    }
    float s = 0.f;
#pragma unroll
    for (int j = 0; j < KP_N; ++j) s += acc[j];
    float mu = s * (1.f / KP_N);
    float sq = 0.f;
#pragma unroll
    for (int j = 0; j < KP_N; ++j) { float d = acc[j] - mu; sq += d * d; }
    float inv = rsqrtf(sq * (1.f / KP_N) + EPS);
#pragma unroll
    for (int j = 0; j < KP_N; ++j)
        npj[(long)n * KP_N + j] = fmaxf((acc[j] - mu) * inv * g1[j] + b1[j], 0.f);
}

// ---------------- generic WMMA GEMM: C[M,N] = A[M,K](f32) @ Bt[N,K](f16) + bias ----------------
// block = 256 threads = 8 waves; blockIdx.x = 16-row tile; waves stride over N/16 tiles.
// A tile async-copied global->LDS (ASYNCcnt), then swizzled once into f16 fragment order
// so the hot loop reads each v16h A fragment as two ds_load_b128.

__global__ __launch_bounds__(256) void wmma_gemm_kernel(
    const float* __restrict__ A, const _Float16* __restrict__ Bt,
    const float* __restrict__ bias, float* __restrict__ C, int N, int K) {
    __shared__ float    s_a[16 * 256];          // raw f32 A tile (K <= 256)
    __shared__ _Float16 s_a16[8 * 32 * 16];     // swizzled f16 fragments (ks, lane, i)
    const int tid = threadIdx.x;
    const int lane = tid & 31;
    const int wave = tid >> 5;
    const int m0 = blockIdx.x * 16;

    // ---- async stage of contiguous A tile: A[m0*K .. (m0+16)*K) -> LDS ----
    {
        const int nvec = (16 * K) >> 2;  // b128 transfers (exact for K=128/256)
        const float* gbase = A + (long)m0 * K;
        for (int t = tid; t < nvec; t += 256) {
            unsigned lds_off = (unsigned)(size_t)(&s_a[t * 4]);
            unsigned long long ga = (unsigned long long)(gbase + (long)t * 4);
            asm volatile("global_load_async_to_lds_b128 %0, %1, off"
                         :: "v"(lds_off), "v"(ga) : "memory");
        }
        asm volatile("s_wait_asynccnt 0x0" ::: "memory");
    }
    __syncthreads();

    // ---- swizzle+convert: (m,k) -> fragment order; each wave handles 2 rows ----
    for (int mm = 0; mm < 2; ++mm) {
        int m = wave * 2 + mm;
        for (int k = lane; k < K; k += 32) {
            int ks = k >> 5, krem = k & 31;
            int hi = (krem >> 4) & 1;
            int i = krem & 15;
            int lane2 = m + (hi << 4);
            s_a16[(ks * 32 + lane2) * 16 + i] = (_Float16)s_a[m * K + k];
        }
    }
    __syncthreads();

    const int ksteps = K >> 5;
    for (int nt = wave; nt < (N >> 4); nt += 8) {
        const int ncol = nt * 16 + (lane & 15);
        const _Float16* brow = Bt + (long)ncol * K + (lane & 16);
        if (nt + 8 < (N >> 4))
            __builtin_prefetch(Bt + (long)(nt + 8) * 16 * K + (lane & 15) * K, 0, 1);
        v8f acc = {};
        for (int ks = 0; ks < ksteps; ++ks) {
            v16h a = *(const v16h*)&s_a16[(ks * 32 + lane) * 16];
            v16h b = *(const v16h*)(brow + (ks << 5));
            acc = __builtin_amdgcn_wmma_f32_16x16x32_f16(false, a, false, b,
                                                         (short)0, acc, false, false);
        }
        const int hi8 = (lane & 16) >> 1;
#pragma unroll
        for (int r = 0; r < 8; ++r)
            C[(long)(m0 + r + hi8) * N + ncol] = acc[r] + bias[ncol];
    }
}

// ---------------- kron edge GEMM ----------------
// block = 128 threads = 4 waves; each wave owns 16 edges. The 16x416 outer-product A tile
// is built ONCE per wave in swizzled f16 fragment order; 8 n-tile accumulators live in
// registers; per-edge LN done with cross-lane shuffles; scatter via fp atomics.

__global__ __launch_bounds__(128) void kron_edge_kernel(
    const float* __restrict__ npj, const _Float16* __restrict__ Wk2T,
    const float* __restrict__ bk2, const float* __restrict__ g2,
    const float* __restrict__ b2, const int* __restrict__ src,
    const int* __restrict__ dst, float* __restrict__ kron_feat) {
    __shared__ float    s_npj[4][16][2 * KP_N];            // 10 KB
    __shared__ _Float16 s_a16[4][KRON_KS * 32 * 16];       // 52 KB swizzled kron-A
    const int lane = threadIdx.x & 31;
    const int wave = threadIdx.x >> 5;
    const int ebase = (blockIdx.x * 4 + wave) * 16;

    // async gather of src/dst npj vectors (16 edges x 40 floats per wave)
    for (int t = lane; t < 16 * 2 * KP_N; t += 32) {
        int e = t / (2 * KP_N), f = t % (2 * KP_N);
        int node = (f < KP_N) ? src[ebase + e] : dst[ebase + e];
        int j = (f < KP_N) ? f : f - KP_N;
        unsigned lds_off = (unsigned)(size_t)(&s_npj[wave][e][f]);
        unsigned long long ga = (unsigned long long)(npj + (long)node * KP_N + j);
        asm volatile("global_load_async_to_lds_b32 %0, %1, off"
                     :: "v"(lds_off), "v"(ga) : "memory");
    }
    asm volatile("s_wait_asynccnt 0x0" ::: "memory");
    __syncthreads();

    // build swizzled f16 A tile once: A[m][k] = npj_src[m][k/20] * npj_dst[m][k%20]
    for (int m = 0; m < 16; ++m) {
        float* row = &s_npj[wave][m][0];
        for (int k = lane; k < KRON_KP; k += 32) {
            float av = 0.f;
            if (k < KRON_K) {
                int ii = k / KP_N;
                int jj = k - ii * KP_N;
                av = row[ii] * row[KP_N + jj];
            }
            int ks = k >> 5, krem = k & 31;
            int hi = (krem >> 4) & 1;
            int i = krem & 15;
            int lane2 = m + (hi << 4);
            s_a16[wave][(ks * 32 + lane2) * 16 + i] = (_Float16)av;
        }
    }
    __syncthreads();

    // GEMM: ks outer so each A fragment feeds all 8 n-tiles
    v8f accs[8];
#pragma unroll
    for (int nt = 0; nt < 8; ++nt) accs[nt] = (v8f){};
    const _Float16* bbase = Wk2T + (lane & 15) * KRON_KP + (lane & 16);
    for (int ks = 0; ks < KRON_KS; ++ks) {
        v16h a = *(const v16h*)&s_a16[wave][(ks * 32 + lane) * 16];
#pragma unroll
        for (int nt = 0; nt < 8; ++nt) {
            v16h b = *(const v16h*)(bbase + (long)nt * 16 * KRON_KP + (ks << 5));
            accs[nt] = __builtin_amdgcn_wmma_f32_16x16x32_f16(false, a, false, b,
                                                              (short)0, accs[nt], false, false);
        }
    }

    // per-edge LN (128 cols spread over 16 lanes x 8 nt) + relu + atomic scatter
    const int colbase = lane & 15;
    const int hi8 = (lane & 16) >> 1;
    float g2c[8], b2c[8], bkc[8];
#pragma unroll
    for (int nt = 0; nt < 8; ++nt) {
        int c = nt * 16 + colbase;
        g2c[nt] = g2[c]; b2c[nt] = b2[c]; bkc[nt] = bk2[c];
    }
#pragma unroll
    for (int r = 0; r < 8; ++r) {
        float s = 0.f, q = 0.f;
#pragma unroll
        for (int nt = 0; nt < 8; ++nt) {
            float v = accs[nt][r] + bkc[nt];
            s += v; q += v * v;
        }
#pragma unroll
        for (int o = 1; o < 16; o <<= 1) { s += __shfl_xor(s, o); q += __shfl_xor(q, o); }
        float mu = s * (1.f / D_N);
        float inv = rsqrtf(q * (1.f / D_N) - mu * mu + EPS);
        int d = dst[ebase + r + hi8];
#pragma unroll
        for (int nt = 0; nt < 8; ++nt) {
            float y = (accs[nt][r] + bkc[nt] - mu) * inv * g2c[nt] + b2c[nt];
            y = fmaxf(y, 0.f);
            atomicAdd(&kron_feat[(long)d * D_N + nt * 16 + colbase], y);
        }
    }
}

// ---------------- attention branch ----------------

__global__ void edge_logit_kernel(const float* __restrict__ nf, const int* __restrict__ src,
                                  const int* __restrict__ dst, const float* __restrict__ We,
                                  const float* __restrict__ be, float* __restrict__ logits,
                                  unsigned int* __restrict__ lmax_u) {
    int e = blockIdx.x * blockDim.x + threadIdx.x;
    if (e >= E_N) return;
    const float* xd = nf + (long)dst[e] * D_N;
    const float* xs = nf + (long)src[e] * D_N;
    float s = be[0];
    for (int i = 0; i < D_N; ++i) s += xd[i] * We[i];
    for (int i = 0; i < D_N; ++i) s += xs[i] * We[D_N + i];
    s = fmaxf(s, 0.f);
    logits[e] = s;
    atomicMax(&lmax_u[dst[e]], __float_as_uint(s));  // valid: s >= 0
}

__global__ void edge_exp_kernel(const float* __restrict__ logits, const int* __restrict__ dst,
                                const float* __restrict__ lmax, float* __restrict__ a,
                                float* __restrict__ asum) {
    int e = blockIdx.x * blockDim.x + threadIdx.x;
    if (e >= E_N) return;
    int d = dst[e];
    float v = expf(logits[e] - lmax[d]);
    a[e] = v;
    atomicAdd(&asum[d], v);
}

__global__ __launch_bounds__(128) void ctx_scatter_kernel(
    const float* __restrict__ hv, const float* __restrict__ a, const float* __restrict__ asum,
    const int* __restrict__ src, const int* __restrict__ dst, float* __restrict__ context) {
    int e = blockIdx.x;
    int c = threadIdx.x;
    int d = dst[e];
    float w = a[e] / asum[d];
    atomicAdd(&context[(long)d * D_N + c], hv[(long)src[e] * D_N + c] * w);
}

// ---------------- GRU gates + relu + LN -> cat[:, :128] ----------------

__global__ __launch_bounds__(128) void gru_ln_kernel(
    const float* __restrict__ gi, const float* __restrict__ gh, const float* __restrict__ nf,
    const float* __restrict__ g, const float* __restrict__ b, float* __restrict__ cat) {
    int n = blockIdx.x, c = threadIdx.x;
    long base = (long)n * 384;
    float r = 1.f / (1.f + expf(-(gi[base + c] + gh[base + c])));
    float z = 1.f / (1.f + expf(-(gi[base + 128 + c] + gh[base + 128 + c])));
    float nn = tanhf(gi[base + 256 + c] + r * gh[base + 256 + c]);
    float h = (1.f - z) * nn + z * nf[(long)n * D_N + c];
    h = fmaxf(h, 0.f);
    float s = h, sq = h * h;
#pragma unroll
    for (int o = 16; o > 0; o >>= 1) { s += __shfl_xor(s, o); sq += __shfl_xor(sq, o); }
    __shared__ float ps[4], pq[4];
    int w = threadIdx.x >> 5, lane = threadIdx.x & 31;
    if (lane == 0) { ps[w] = s; pq[w] = sq; }
    __syncthreads();
    float ts = ps[0] + ps[1] + ps[2] + ps[3];
    float tq = pq[0] + pq[1] + pq[2] + pq[3];
    float mu = ts * (1.f / D_N);
    float inv = rsqrtf(tq * (1.f / D_N) - mu * mu + EPS);
    cat[(long)n * 256 + c] = (h - mu) * inv * g[c] + b[c];
}

__global__ void cat_kron_kernel(const float* __restrict__ kf, float* __restrict__ cat, long n) {
    long i = (long)blockIdx.x * blockDim.x + threadIdx.x;
    if (i >= n) return;
    long node = i >> 7;
    int c = (int)(i & 127);
    cat[node * 256 + 128 + c] = kf[i];
}

__global__ __launch_bounds__(128) void final_ln_kernel(
    const float* __restrict__ y, const float* __restrict__ g, const float* __restrict__ b,
    float* __restrict__ out) {
    int n = blockIdx.x, c = threadIdx.x;
    float h = y[(long)n * D_N + c];
    float s = h, sq = h * h;
#pragma unroll
    for (int o = 16; o > 0; o >>= 1) { s += __shfl_xor(s, o); sq += __shfl_xor(sq, o); }
    __shared__ float ps[4], pq[4];
    int w = threadIdx.x >> 5, lane = threadIdx.x & 31;
    if (lane == 0) { ps[w] = s; pq[w] = sq; }
    __syncthreads();
    float ts = ps[0] + ps[1] + ps[2] + ps[3];
    float tq = pq[0] + pq[1] + pq[2] + pq[3];
    float mu = ts * (1.f / D_N);
    float inv = rsqrtf(tq * (1.f / D_N) - mu * mu + EPS);
    out[(long)n * D_N + c] = fmaxf((h - mu) * inv * g[c] + b[c], 0.f);
}

// ---------------- launcher ----------------

extern "C" void kernel_launch(void* const* d_in, const int* in_sizes, int n_in,
                              void* d_out, int out_size, void* d_ws, size_t ws_size,
                              hipStream_t stream) {
    const float* nf     = (const float*)d_in[0];
    const int*   src    = (const int*)d_in[1];
    const int*   dst    = (const int*)d_in[2];
    const float* W_edge = (const float*)d_in[3];
    const float* b_edge = (const float*)d_in[4];
    const float* W_pn   = (const float*)d_in[5];
    const float* b_pn   = (const float*)d_in[6];
    const float* W_ih   = (const float*)d_in[7];
    const float* b_ih   = (const float*)d_in[8];
    const float* W_hh   = (const float*)d_in[9];
    const float* b_hh   = (const float*)d_in[10];
    const float* ln_g   = (const float*)d_in[11];
    const float* ln_b   = (const float*)d_in[12];
    const float* Wk1    = (const float*)d_in[13];
    const float* bk1    = (const float*)d_in[14];
    const float* lnk1_g = (const float*)d_in[15];
    const float* lnk1_b = (const float*)d_in[16];
    const float* Wk2    = (const float*)d_in[17];
    const float* bk2    = (const float*)d_in[18];
    const float* lnk2_g = (const float*)d_in[19];
    const float* lnk2_b = (const float*)d_in[20];
    const float* Wc     = (const float*)d_in[21];
    const float* bc     = (const float*)d_in[22];
    const float* lnc_g  = (const float*)d_in[23];
    const float* lnc_b  = (const float*)d_in[24];

    // ---- workspace layout (accumulation buffers first, contiguous, zeroed each call) ----
    char* p = (char*)d_ws;
    float* context   = (float*)p; p += (size_t)V_N * 128 * 4;
    float* asum      = (float*)p; p += (size_t)V_N * 4;
    float* lmax      = (float*)p; p += (size_t)V_N * 4;
    float* kron_feat = (float*)p; p += (size_t)V_N * 128 * 4;
    const long nzero = (long)V_N * 258;   // context + asum + lmax + kron_feat
    float* npj    = (float*)p; p += (size_t)V_N * KP_N * 4;
    float* hv     = (float*)p; p += (size_t)V_N * 128 * 4;
    float* gh     = (float*)p; p += (size_t)V_N * 384 * 4;
    float* gi     = (float*)p; p += (size_t)V_N * 384 * 4;
    float* logits = (float*)p; p += (size_t)E_N * 4;
    float* aexp   = (float*)p; p += (size_t)E_N * 4;
    float* cat    = (float*)p; p += (size_t)V_N * 256 * 4;
    float* ybuf   = (float*)p; p += (size_t)V_N * 128 * 4;
    _Float16* Wk2T  = (_Float16*)p; p += (size_t)128 * KRON_KP * 2;
    _Float16* WpnT  = (_Float16*)p; p += (size_t)128 * 128 * 2;
    _Float16* Whh16 = (_Float16*)p; p += (size_t)384 * 128 * 2;
    _Float16* Wih16 = (_Float16*)p; p += (size_t)384 * 128 * 2;
    _Float16* WcT   = (_Float16*)p; p += (size_t)128 * 256 * 2;

    float* out = (float*)d_out;

    // 0) zero accumulation buffers (deterministic across replays)
    zero_kernel<<<(unsigned)((nzero + 255) / 256), 256, 0, stream>>>((float*)d_ws, nzero);

    // 1) weight conversions to f16 (B stored transposed: [N][Kpad])
    {
        long n;
        n = (long)128 * KRON_KP;
        cvt_wt_kernel<<<(unsigned)((n + 255) / 256), 256, 0, stream>>>(Wk2T, Wk2, KRON_K, 128, KRON_KP, 1);
        n = (long)128 * 128;
        cvt_wt_kernel<<<(unsigned)((n + 255) / 256), 256, 0, stream>>>(WpnT, W_pn, 128, 128, 128, 1);
        n = (long)384 * 128;
        cvt_wt_kernel<<<(unsigned)((n + 255) / 256), 256, 0, stream>>>(Whh16, W_hh, 128, 384, 128, 0);
        cvt_wt_kernel<<<(unsigned)((n + 255) / 256), 256, 0, stream>>>(Wih16, W_ih, 128, 384, 128, 0);
        n = (long)128 * 256;
        cvt_wt_kernel<<<(unsigned)((n + 255) / 256), 256, 0, stream>>>(WcT, Wc, 256, 128, 256, 1);
    }

    // 2) npj = relu(LN(nf @ Wk1 + bk1))
    npj_kernel<<<(V_N + 255) / 256, 256, 0, stream>>>(nf, Wk1, bk1, lnk1_g, lnk1_b, npj);

    // 3) hv = nf @ W_pn + b_pn ; gh = nf @ W_hh^T + b_hh   (WMMA, async A stage)
    wmma_gemm_kernel<<<V_N / 16, 256, 0, stream>>>(nf, WpnT, b_pn, hv, 128, 128);
    wmma_gemm_kernel<<<V_N / 16, 256, 0, stream>>>(nf, Whh16, b_hh, gh, 384, 128);

    // 4) kron branch: per-edge outer product GEMM + LN + relu + scatter (WMMA)
    kron_edge_kernel<<<E_N / 64, 128, 0, stream>>>(npj, Wk2T, bk2, lnk2_g, lnk2_b, src, dst, kron_feat);

    // 5) attention: logits + segment max, exp + segment sum, weighted scatter
    edge_logit_kernel<<<(E_N + 255) / 256, 256, 0, stream>>>(nf, src, dst, W_edge, b_edge, logits,
                                                             (unsigned int*)lmax);
    edge_exp_kernel<<<(E_N + 255) / 256, 256, 0, stream>>>(logits, dst, lmax, aexp, asum);
    ctx_scatter_kernel<<<E_N, 128, 0, stream>>>(hv, aexp, asum, src, dst, context);
    relu_kernel<<<(unsigned)(((long)V_N * 128 + 255) / 256), 256, 0, stream>>>(context, (long)V_N * 128);

    // 6) gi = context @ W_ih^T + b_ih   (WMMA, async A stage)
    wmma_gemm_kernel<<<V_N / 16, 256, 0, stream>>>(context, Wih16, b_ih, gi, 384, 128);

    // 7) GRU gates + relu + LN -> cat[:, :128]; kron_feat -> cat[:, 128:]
    gru_ln_kernel<<<V_N, 128, 0, stream>>>(gi, gh, nf, ln_g, ln_b, cat);
    cat_kron_kernel<<<(unsigned)(((long)V_N * 128 + 255) / 256), 256, 0, stream>>>(kron_feat, cat,
                                                                                  (long)V_N * 128);

    // 8) y = cat @ Wc + bc (WMMA, async A stage); out = relu(LN(y))
    wmma_gemm_kernel<<<V_N / 16, 256, 0, stream>>>(cat, WcT, bc, ybuf, 128, 256);
    final_ln_kernel<<<V_N, 128, 0, stream>>>(ybuf, lnc_g, lnc_b, out);
}
